// Grapher_35493609734670
// MI455X (gfx1250) — matrix operand order repaired and checked by reading, not compiled, not run on hardware
//
#include <hip/hip_runtime.h>
#include <hip/hip_bf16.h>

// ---------------------------------------------------------------------------
// CDNA5 (gfx1250): conv3x3+BN -> strided self-attention -> concat+conv1x1+
// BN+GELU -> conv3x3+BN -> +x.  All GEMMs via v_wmma_f32_16x16x32_bf16
// (wave32, f32 accumulate).  Activation tiles staged to LDS with the gfx1250
// async path: global_load_async_to_lds_b128 + s_wait_asynccnt (ASYNCcnt).
// ---------------------------------------------------------------------------

typedef __attribute__((ext_vector_type(16))) __bf16 v16bf;
typedef __attribute__((ext_vector_type(8)))  float  v8f;

union FragB16 {
    v16bf v;
    uint4 u[2];
    unsigned short s[16];
};

#define BB   16
#define CC   64
#define HH   128
#define WW   128
#define C2   128
#define HP   130
#define WP   130
#define NTOK 1024
#define EPSV 1e-5f

// Set to 0 to fall back to plain vector-load + ds-store staging.
#define USE_ASYNC_LDS 1

__device__ __forceinline__ unsigned short f2bf(float f) {
    union { float f; unsigned u; } c; c.f = f;
    unsigned r = c.u + 0x7fffu + ((c.u >> 16) & 1u);   // round-to-nearest-even
    return (unsigned short)(r >> 16);
}
__device__ __forceinline__ unsigned pk2(unsigned short a, unsigned short b) {
    return (unsigned)a | ((unsigned)b << 16);
}
__device__ __forceinline__ v8f vzero8() {
    v8f z;
#pragma unroll
    for (int i = 0; i < 8; ++i) z[i] = 0.f;
    return z;
}
#define WMMA_BF16(A, Bv, Cv) \
    __builtin_amdgcn_wmma_f32_16x16x32_bf16(false, (A), false, (Bv), (short)0, (Cv), false, false)

// 16-byte global -> LDS async copy (per-lane addresses), ISA 15.18 VGLOBAL op 98.
__device__ __forceinline__ void async_copy16(void* lds, const void* g) {
#if USE_ASYNC_LDS
    unsigned loff = (unsigned)(uintptr_t)lds;                 // LDS byte address
    unsigned long long ga = (unsigned long long)(uintptr_t)g; // 64-bit global addr
    asm volatile("global_load_async_to_lds_b128 %0, %1, off"
                 :: "v"(loff), "v"(ga)
                 : "memory");
#else
    *(uint4*)lds = *(const uint4*)g;
#endif
}
__device__ __forceinline__ void async_join() {
#if USE_ASYNC_LDS
    asm volatile("s_wait_asynccnt 0" ::: "memory");
#endif
}

// ---------------------------------------------------------------------------
// Weight prep: fold BN scale into conv weights, bf16, row-major [M][K].
// ---------------------------------------------------------------------------
__global__ void k_prep_w1(const float* w, const float* bias, const float* g,
                          const float* be, const float* m, const float* v,
                          unsigned short* w1, float* b1) {
    int tid = blockIdx.x * blockDim.x + threadIdx.x;
    if (tid < CC) {
        float s = g[tid] * rsqrtf(v[tid] + EPSV);
        b1[tid] = (bias[tid] - m[tid]) * s + be[tid];
    }
    if (tid >= CC * 576) return;
    int oc = tid / 576, k = tid % 576;
    int tap = k / CC, ic = k % CC;                    // k = tap*64 + ic
    float s = g[oc] * rsqrtf(v[oc] + EPSV);
    w1[tid] = f2bf(w[(oc * CC + ic) * 9 + tap] * s);
}

__global__ void k_prep_wg(const float* w, const float* bias, const float* g,
                          const float* be, const float* m, const float* v,
                          unsigned short* wg, float* bg) {
    int tid = blockIdx.x * blockDim.x + threadIdx.x;
    if (tid < C2) {
        float s = g[tid] * rsqrtf(v[tid] + EPSV);
        bg[tid] = (bias[tid] - m[tid]) * s + be[tid];
    }
    if (tid >= C2 * C2) return;
    int oc = tid / C2;
    float s = g[oc] * rsqrtf(v[oc] + EPSV);
    wg[tid] = f2bf(w[tid] * s);                        // gc_w is [oc][ic][1][1]
}

__global__ void k_prep_w2(const float* w, const float* bias, const float* g,
                          const float* be, const float* m, const float* v,
                          unsigned short* w2, float* b2) {
    int tid = blockIdx.x * blockDim.x + threadIdx.x;
    if (tid < CC) {
        float s = g[tid] * rsqrtf(v[tid] + EPSV);
        b2[tid] = (bias[tid] - m[tid]) * s + be[tid];
    }
    if (tid >= CC * 1152) return;
    int oc = tid / 1152, k = tid % 1152;
    int tap = k / C2, ic = k % C2;                     // k = tap*128 + ic
    float s = g[oc] * rsqrtf(v[oc] + EPSV);
    w2[tid] = f2bf(w[(oc * C2 + ic) * 9 + tap] * s);
}

// ---------------------------------------------------------------------------
// NCHW f32 -> padded NHWC bf16 (zero halo) for the implicit-GEMM conv1.
// ---------------------------------------------------------------------------
__global__ void k_xpad(const float* __restrict__ x, unsigned short* __restrict__ xp) {
    int tid = blockIdx.x * blockDim.x + threadIdx.x;
    if (tid >= BB * HP * WP * CC) return;
    int ic = tid & 63;
    int r  = tid >> 6;
    int gx = r % WP; r /= WP;
    int gy = r % HP; int b = r / HP;
    float val = 0.f;
    if (gy >= 1 && gy <= HH && gx >= 1 && gx <= WW)
        val = x[(((size_t)b * CC + ic) * HH + (gy - 1)) * WW + (gx - 1)];
    xp[tid] = f2bf(val);
}

// ---------------------------------------------------------------------------
// conv1: 3x3, 64->64. Block = (b,y) row of 128 px; 3-row halo strip staged to
// LDS via async copy (49,920 B, contiguous in padded NHWC); wave = 16oc x 64px.
// ---------------------------------------------------------------------------
__global__ void __launch_bounds__(256) k_conv1(const unsigned short* __restrict__ xp,
                                               const unsigned short* __restrict__ w1,
                                               const float* __restrict__ b1,
                                               unsigned short* __restrict__ hbuf) {
    __shared__ __align__(16) unsigned short xls[3 * WP * CC];   // 49,920 B
    int b    = blockIdx.x >> 7;
    int y    = blockIdx.x & 127;
    int tid  = threadIdx.x;
    int wave = tid >> 5;
    int lane = tid & 31;
    int ln = lane & 15, half = lane >> 4;
    int mt = wave & 3, grp = wave >> 2;
    int oc0 = mt * 16;

    {   // stage rows y..y+2 of the padded input (one contiguous region)
        const char* src = (const char*)(xp + ((size_t)b * HP + y) * WP * CC);
        char* dst = (char*)xls;
        for (int c = tid; c < (3 * WP * CC * 2) / 16; c += 256)
            async_copy16(dst + c * 16, src + c * 16);
        async_join();
    }
    __syncthreads();

    v8f acc[4];
#pragma unroll
    for (int t = 0; t < 4; ++t) acc[t] = vzero8();

    for (int tap = 0; tap < 9; ++tap) {
        int dy = tap / 3, dx = tap % 3;
#pragma unroll
        for (int ck = 0; ck < 2; ++ck) {
            FragB16 a;
            const unsigned short* wp = w1 + (size_t)(oc0 + ln) * 576 + tap * 64 + ck * 32 + 16 * half;
            a.u[0] = *(const uint4*)(wp);
            a.u[1] = *(const uint4*)(wp + 8);
            int icb = ck * 32 + 16 * half;
#pragma unroll
            for (int t = 0; t < 4; ++t) {
                int px = grp * 64 + t * 16 + ln;
                const unsigned short* bp = xls + ((dy * WP + px + dx) * CC + icb);
                FragB16 bb;
                bb.u[0] = ((const uint4*)bp)[0];
                bb.u[1] = ((const uint4*)bp)[1];
                acc[t] = WMMA_BF16(a.v, bb.v, acc[t]);
            }
        }
    }
    // epilogue: +bias(folded BN1), store NHWC bf16 (8 contiguous channels/lane)
    int ocb = oc0 + 8 * half;
    float bs[8];
#pragma unroll
    for (int r = 0; r < 8; ++r) bs[r] = b1[ocb + r];
#pragma unroll
    for (int t = 0; t < 4; ++t) {
        int px = grp * 64 + t * 16 + ln;
        unsigned short h[8];
#pragma unroll
        for (int r = 0; r < 8; ++r) h[r] = f2bf(acc[t][r] + bs[r]);
        uint4 o;
        o.x = pk2(h[0], h[1]); o.y = pk2(h[2], h[3]);
        o.z = pk2(h[4], h[5]); o.w = pk2(h[6], h[7]);
        *(uint4*)(hbuf + (((size_t)b * HH + y) * WW + px) * CC + ocb) = o;
    }
}

// ---------------------------------------------------------------------------
// q (token-major) and qT (channel-major) from h[:, :, ::4, ::4].
// ---------------------------------------------------------------------------
__global__ void k_qprep(const unsigned short* __restrict__ hbuf,
                        unsigned short* __restrict__ q,
                        unsigned short* __restrict__ qT) {
    int tid = blockIdx.x * blockDim.x + threadIdx.x;
    if (tid >= BB * NTOK * CC) return;
    int c = tid & 63;
    int t = (tid >> 6) & 1023;
    int b = tid >> 16;
    int ty = t >> 5, tx = t & 31;
    unsigned short v = hbuf[(((size_t)b * HH + ty * 4) * WW + tx * 4) * CC + c];
    q[tid] = v;
    qT[((size_t)b * CC + c) * NTOK + t] = v;
}

// ---------------------------------------------------------------------------
// Attention: per wave 16 query rows; sweep1 row-max, sweep2 recompute S,
// exp, stage P through LDS as A-fragment, accumulate o = P @ q via WMMA.
// ---------------------------------------------------------------------------
__global__ void __launch_bounds__(256) k_att(const unsigned short* __restrict__ q,
                                             const unsigned short* __restrict__ qT,
                                             unsigned short* __restrict__ osm) {
    __shared__ __align__(16) unsigned short Pls[8][16][32];
    int b    = blockIdx.x >> 3;
    int rg   = blockIdx.x & 7;
    int wave = threadIdx.x >> 5;
    int lane = threadIdx.x & 31;
    int ln = lane & 15, half = lane >> 4;
    int m0 = (rg * 8 + wave) * 16;
    const float scale = 0.125f;                         // 64^-0.5

    const unsigned short* qb  = q  + (size_t)b * NTOK * CC;
    const unsigned short* qtb = qT + (size_t)b * CC * NTOK;

    FragB16 A0, A1;                                     // 16 q-rows, K=64 (2 chunks)
    {
        const unsigned short* ap = qb + (size_t)(m0 + ln) * CC;
        A0.u[0] = *(const uint4*)(ap + 16 * half);
        A0.u[1] = *(const uint4*)(ap + 8 + 16 * half);
        A1.u[0] = *(const uint4*)(ap + 32 + 16 * half);
        A1.u[1] = *(const uint4*)(ap + 40 + 16 * half);
    }

    float rmax[8];
#pragma unroll
    for (int r = 0; r < 8; ++r) rmax[r] = -3.0e38f;

    // ---- sweep 1: row max ----
    for (int ct = 0; ct < 32; ++ct) {
#pragma unroll
        for (int nt = 0; nt < 2; ++nt) {
            int tc = ct * 32 + nt * 16 + ln;
            const unsigned short* bp = qb + (size_t)tc * CC;
            FragB16 B0, B1;
            B0.u[0] = ((const uint4*)(bp + 16 * half))[0];
            B0.u[1] = ((const uint4*)(bp + 16 * half))[1];
            B1.u[0] = ((const uint4*)(bp + 32 + 16 * half))[0];
            B1.u[1] = ((const uint4*)(bp + 32 + 16 * half))[1];
            v8f s = vzero8();
            s = WMMA_BF16(A0.v, B0.v, s);
            s = WMMA_BF16(A1.v, B1.v, s);
#pragma unroll
            for (int r = 0; r < 8; ++r) rmax[r] = fmaxf(rmax[r], s[r] * scale);
        }
    }
#pragma unroll
    for (int r = 0; r < 8; ++r)
#pragma unroll
        for (int msk = 1; msk < 16; msk <<= 1)
            rmax[r] = fmaxf(rmax[r], __shfl_xor(rmax[r], msk, 32));

    // ---- sweep 2: recompute, exp, accumulate o ----
    float rsum[8];
    v8f oacc[4];
#pragma unroll
    for (int r = 0; r < 8; ++r) rsum[r] = 0.f;
#pragma unroll
    for (int nc = 0; nc < 4; ++nc) oacc[nc] = vzero8();

    for (int ct = 0; ct < 32; ++ct) {
#pragma unroll
        for (int nt = 0; nt < 2; ++nt) {
            int tc = ct * 32 + nt * 16 + ln;
            const unsigned short* bp = qb + (size_t)tc * CC;
            FragB16 B0, B1;
            B0.u[0] = ((const uint4*)(bp + 16 * half))[0];
            B0.u[1] = ((const uint4*)(bp + 16 * half))[1];
            B1.u[0] = ((const uint4*)(bp + 32 + 16 * half))[0];
            B1.u[1] = ((const uint4*)(bp + 32 + 16 * half))[1];
            v8f s = vzero8();
            s = WMMA_BF16(A0.v, B0.v, s);
            s = WMMA_BF16(A1.v, B1.v, s);
#pragma unroll
            for (int r = 0; r < 8; ++r) {
                float p = expf(s[r] * scale - rmax[r]);
                rsum[r] += p;
                Pls[wave][r + 8 * half][nt * 16 + ln] = f2bf(p);
            }
        }
        __syncthreads();
        FragB16 AP;                                     // P tile as A-fragment
        AP.u[0] = *(const uint4*)&Pls[wave][ln][16 * half];
        AP.u[1] = *(const uint4*)&Pls[wave][ln][8 + 16 * half];
#pragma unroll
        for (int nc = 0; nc < 4; ++nc) {
            int ch = nc * 16 + ln;
            const unsigned short* bp = qtb + (size_t)ch * NTOK + ct * 32 + 16 * half;
            FragB16 Bq;
            Bq.u[0] = ((const uint4*)bp)[0];
            Bq.u[1] = ((const uint4*)bp)[1];
            oacc[nc] = WMMA_BF16(AP.v, Bq.v, oacc[nc]);
        }
        __syncthreads();
    }

#pragma unroll
    for (int r = 0; r < 8; ++r)
#pragma unroll
        for (int msk = 1; msk < 16; msk <<= 1)
            rsum[r] += __shfl_xor(rsum[r], msk, 32);

#pragma unroll
    for (int nc = 0; nc < 4; ++nc) {
        int ch = nc * 16 + ln;
#pragma unroll
        for (int r = 0; r < 8; ++r) {
            int tok = m0 + r + 8 * half;
            osm[((size_t)b * NTOK + tok) * CC + ch] = f2bf(oacc[nc][r] / rsum[r]);
        }
    }
}

// ---------------------------------------------------------------------------
// gc: 1x1 conv over virtual concat([h, upsample4(o)]) -> GELU -> gpad interior.
// 64-px activation tile (+16 attention tokens) staged to LDS via async copy;
// all 8 M-tiles (waves) share it.  K chunks 0-1 from h, 2-3 from o.
// ---------------------------------------------------------------------------
__global__ void __launch_bounds__(256) k_gc(const unsigned short* __restrict__ hbuf,
                                            const unsigned short* __restrict__ osm,
                                            const unsigned short* __restrict__ wg,
                                            const float* __restrict__ bg,
                                            unsigned short* __restrict__ gpad) {
    __shared__ __align__(16) unsigned short hls[64 * CC];   // 8 KB
    __shared__ __align__(16) unsigned short ols[16 * CC];   // 2 KB
    int b    = blockIdx.x >> 8;
    int pg   = blockIdx.x & 255;
    int tid  = threadIdx.x;
    int wave = tid >> 5;
    int lane = tid & 31;
    int ln = lane & 15, half = lane >> 4;
    int oc0 = wave * 16;

    int y0 = (pg * 64) >> 7;            // single row per 64-px group
    int x0 = (pg & 1) * 64;
    {
        const char* hsrc = (const char*)(hbuf + (((size_t)b * HH * WW) + pg * 64) * CC);
        for (int c = tid; c < 512; c += 256)                       // 8192 B
            async_copy16((char*)hls + c * 16, hsrc + c * 16);
        int tok0 = (y0 >> 2) * 32 + (x0 >> 2);
        const char* osrc = (const char*)(osm + ((size_t)b * NTOK + tok0) * CC);
        for (int c = tid; c < 128; c += 256)                       // 2048 B
            async_copy16((char*)ols + c * 16, osrc + c * 16);
        async_join();
    }
    __syncthreads();

    v8f acc[4];
#pragma unroll
    for (int t = 0; t < 4; ++t) acc[t] = vzero8();

#pragma unroll
    for (int ck = 0; ck < 4; ++ck) {
        FragB16 a;
        const unsigned short* wp = wg + (size_t)(oc0 + ln) * C2 + ck * 32 + 16 * half;
        a.u[0] = *(const uint4*)wp;
        a.u[1] = *(const uint4*)(wp + 8);
#pragma unroll
        for (int t = 0; t < 4; ++t) {
            int lx = t * 16 + ln;
            const unsigned short* bp = (ck < 2)
                ? hls + lx * CC + ck * 32 + 16 * half
                : ols + (lx >> 2) * CC + (ck - 2) * 32 + 16 * half;
            FragB16 bb;
            bb.u[0] = ((const uint4*)bp)[0];
            bb.u[1] = ((const uint4*)bp)[1];
            acc[t] = WMMA_BF16(a.v, bb.v, acc[t]);
        }
    }
    int ocb = oc0 + 8 * half;
#pragma unroll
    for (int t = 0; t < 4; ++t) {
        int lx = t * 16 + ln;
        unsigned short h[8];
#pragma unroll
        for (int r = 0; r < 8; ++r) {
            float v = acc[t][r] + bg[ocb + r];
            v = 0.5f * v * (1.f + erff(v * 0.70710678118654752f));   // exact GELU
            h[r] = f2bf(v);
        }
        uint4 o;
        o.x = pk2(h[0], h[1]); o.y = pk2(h[2], h[3]);
        o.z = pk2(h[4], h[5]); o.w = pk2(h[6], h[7]);
        *(uint4*)(gpad + (((size_t)b * HP + 1 + y0) * WP + 1 + x0 + lx) * C2 + ocb) = o;
    }
}

__global__ void k_gborder(unsigned short* __restrict__ gpad) {
    int tid = blockIdx.x * blockDim.x + threadIdx.x;
    if (tid >= BB * HP * WP) return;
    int x = tid % WP;
    int y = (tid / WP) % HP;
    if (y != 0 && y != HP - 1 && x != 0 && x != WP - 1) return;
    uint4 z = make_uint4(0u, 0u, 0u, 0u);
    uint4* p = (uint4*)(gpad + (size_t)tid * C2);
#pragma unroll
    for (int i = 0; i < 16; ++i) p[i] = z;
}

// ---------------------------------------------------------------------------
// fc2: 3x3, 128->64, + bias(folded BN2) + residual x -> f32 NCHW output.
// Block = 64 px; 3x66-px halo strip of gpad staged to LDS via async copy
// (50,688 B); wave = 16oc x 32px (2 tiles).
// ---------------------------------------------------------------------------
__global__ void __launch_bounds__(256) k_conv2(const unsigned short* __restrict__ gpad,
                                               const unsigned short* __restrict__ w2,
                                               const float* __restrict__ b2,
                                               const float* __restrict__ xin,
                                               float* __restrict__ out) {
    __shared__ __align__(16) unsigned short gls[3 * 66 * C2];   // 50,688 B
    int bid  = blockIdx.x;                // 16*128*2 blocks
    int b    = bid >> 8;
    int rem  = bid & 255;
    int y    = rem >> 1;
    int xg   = (rem & 1) * 64;            // padded-col base of the halo window
    int tid  = threadIdx.x;
    int wave = tid >> 5;
    int lane = tid & 31;
    int ln = lane & 15, half = lane >> 4;
    int mt = wave & 3, sub = wave >> 2;   // 16 oc x 32 px per wave
    int oc0 = mt * 16;

    {   // stage 3 row segments of 66 px x 128 ch
        const int SEGC = (66 * C2 * 2) / 16;   // 1056 chunks/segment
        for (int c = tid; c < 3 * SEGC; c += 256) {
            int seg = c / SEGC;
            int off = c - seg * SEGC;
            const char* src = (const char*)(gpad + (((size_t)b * HP + y + seg) * WP + xg) * C2) + off * 16;
            async_copy16((char*)gls + (seg * SEGC + off) * 16, src);
        }
        async_join();
    }
    __syncthreads();

    v8f acc[2];
#pragma unroll
    for (int t = 0; t < 2; ++t) acc[t] = vzero8();

    for (int tap = 0; tap < 9; ++tap) {
        int dy = tap / 3, dx = tap % 3;
#pragma unroll
        for (int ck = 0; ck < 4; ++ck) {
            FragB16 a;
            const unsigned short* wp = w2 + (size_t)(oc0 + ln) * 1152 + tap * 128 + ck * 32 + 16 * half;
            a.u[0] = *(const uint4*)wp;
            a.u[1] = *(const uint4*)(wp + 8);
            int icb = ck * 32 + 16 * half;
#pragma unroll
            for (int t = 0; t < 2; ++t) {
                int lx = sub * 32 + t * 16 + ln;
                const unsigned short* bp = gls + ((dy * 66 + lx + dx) * C2 + icb);
                FragB16 bb;
                bb.u[0] = ((const uint4*)bp)[0];
                bb.u[1] = ((const uint4*)bp)[1];
                acc[t] = WMMA_BF16(a.v, bb.v, acc[t]);
            }
        }
    }
#pragma unroll
    for (int t = 0; t < 2; ++t) {
        int px = xg + sub * 32 + t * 16 + ln;
#pragma unroll
        for (int r = 0; r < 8; ++r) {
            int oc = oc0 + 8 * half + r;
            size_t idx = (((size_t)b * CC + oc) * HH + y) * WW + px;
            out[idx] = acc[t][r] + b2[oc] + xin[idx];
        }
    }
}

// ---------------------------------------------------------------------------
extern "C" void kernel_launch(void* const* d_in, const int* in_sizes, int n_in,
                              void* d_out, int out_size, void* d_ws, size_t ws_size,
                              hipStream_t stream) {
    (void)in_sizes; (void)n_in; (void)out_size; (void)ws_size;
    const float* x     = (const float*)d_in[0];
    const float* fc1_w = (const float*)d_in[1];
    const float* fc1_b = (const float*)d_in[2];
    const float* bn1_g = (const float*)d_in[3];
    const float* bn1_b = (const float*)d_in[4];
    const float* bn1_m = (const float*)d_in[5];
    const float* bn1_v = (const float*)d_in[6];
    const float* gc_w  = (const float*)d_in[7];
    const float* gc_b  = (const float*)d_in[8];
    const float* bng_g = (const float*)d_in[9];
    const float* bng_b = (const float*)d_in[10];
    const float* bng_m = (const float*)d_in[11];
    const float* bng_v = (const float*)d_in[12];
    const float* fc2_w = (const float*)d_in[13];
    const float* fc2_b = (const float*)d_in[14];
    const float* bn2_g = (const float*)d_in[15];
    const float* bn2_b = (const float*)d_in[16];
    const float* bn2_m = (const float*)d_in[17];
    const float* bn2_v = (const float*)d_in[18];
    float* out = (float*)d_out;

    char* ws = (char*)d_ws;
    size_t off = 0;
    auto carve = [&](size_t bytes) -> char* {
        char* p = ws + off;
        off = (off + bytes + 255) & ~(size_t)255;
        return p;
    };
    unsigned short* xpad = (unsigned short*)carve((size_t)BB * HP * WP * CC * 2);   // 34.6 MB
    unsigned short* hbuf = (unsigned short*)carve((size_t)BB * HH * WW * CC * 2);   // 33.6 MB
    unsigned short* q    = (unsigned short*)carve((size_t)BB * NTOK * CC * 2);      //  2.1 MB
    unsigned short* qT   = (unsigned short*)carve((size_t)BB * CC * NTOK * 2);      //  2.1 MB
    unsigned short* osm  = (unsigned short*)carve((size_t)BB * NTOK * CC * 2);      //  2.1 MB
    unsigned short* gpad = (unsigned short*)carve((size_t)BB * HP * WP * C2 * 2);   // 69.2 MB
    unsigned short* w1   = (unsigned short*)carve((size_t)CC * 576 * 2);
    unsigned short* wg   = (unsigned short*)carve((size_t)C2 * C2 * 2);
    unsigned short* w2   = (unsigned short*)carve((size_t)CC * 1152 * 2);
    float*          b1   = (float*)carve(CC * 4);
    float*          bg   = (float*)carve(C2 * 4);
    float*          b2   = (float*)carve(CC * 4);

    dim3 blk(256);
    k_prep_w1<<<(CC * 576 + 255) / 256, blk, 0, stream>>>(fc1_w, fc1_b, bn1_g, bn1_b, bn1_m, bn1_v, w1, b1);
    k_prep_wg<<<(C2 * C2 + 255) / 256, blk, 0, stream>>>(gc_w, gc_b, bng_g, bng_b, bng_m, bng_v, wg, bg);
    k_prep_w2<<<(CC * 1152 + 255) / 256, blk, 0, stream>>>(fc2_w, fc2_b, bn2_g, bn2_b, bn2_m, bn2_v, w2, b2);
    k_xpad<<<(BB * HP * WP * CC + 255) / 256, blk, 0, stream>>>(x, xpad);
    k_conv1<<<BB * HH, blk, 0, stream>>>(xpad, w1, b1, hbuf);
    k_qprep<<<(BB * NTOK * CC + 255) / 256, blk, 0, stream>>>(hbuf, q, qT);
    k_att<<<BB * 8, blk, 0, stream>>>(q, qT, osm);
    k_gborder<<<(BB * HP * WP + 255) / 256, blk, 0, stream>>>(gpad);
    k_gc<<<BB * 256, blk, 0, stream>>>(hbuf, osm, wg, bg, gpad);
    k_conv2<<<BB * HH * 2, blk, 0, stream>>>(gpad, w2, b2, x, out);
}
